// CNN_LSTM_CRF_8916352106580
// MI455X (gfx1250) — matrix operand order repaired and checked by reading, not compiled, not run on hardware
//
#include <hip/hip_runtime.h>
#include <hip/hip_bf16.h>
#include <math.h>

// ---------------------------------------------------------------------------
// CNN-LSTM-CRF forward for MI455X (gfx1250, wave32, WMMA bf16 + async G->LDS)
//   B=8, T=64, L=64, V=50000, E=768, K=256 filters, H=64, N_CLASS=5
// ---------------------------------------------------------------------------

typedef __bf16 bf16_t;
typedef bf16_t v16bf __attribute__((ext_vector_type(16)));
typedef float  v8f   __attribute__((ext_vector_type(8)));

union FragAB {            // one WMMA A/B operand: 16 bf16 per lane = 32 bytes
    v16bf v;
    uint4 q[2];
};

__device__ __forceinline__ unsigned short f2bf(float f) {
    unsigned u = __float_as_uint(f);
    unsigned r = u + 0x7fffu + ((u >> 16) & 1u);   // round-to-nearest-even
    return (unsigned short)(r >> 16);
}

__device__ __forceinline__ float sigm(float x) { return 1.0f / (1.0f + expf(-x)); }

// CDNA5 async DMA: global -> LDS, 16B per lane, tracked by ASYNCcnt
// (CDNA5 ISA: GLOBAL_LOAD_ASYNC_TO_LDS_B128, GV mode)
__device__ __forceinline__ void async_g2l_b128(unsigned lds_off, const void* gaddr) {
    asm volatile("global_load_async_to_lds_b128 %0, %1, off"
                 :: "v"(lds_off), "v"(gaddr)
                 : "memory");
}
__device__ __forceinline__ void wait_async0() {
    asm volatile("s_wait_asynccnt 0x0" ::: "memory");
}

// ---------------------------------------------------------------------------
// 1) Embedding gather + fp32 -> bf16  (X[row][768], row = sentence*64 + pos)
// ---------------------------------------------------------------------------
__global__ void __launch_bounds__(256)
gather_embed_bf16(const int* __restrict__ ids, const float* __restrict__ emb,
                  unsigned short* __restrict__ X, int nrows)
{
    int idx   = blockIdx.x * blockDim.x + threadIdx.x;   // float4 units
    int total = nrows * 192;                             // 768/4 per row
    if (idx >= total) return;
    int row = idx / 192;
    int e4  = idx - row * 192;
    int tok = ids[row];
    const float4 v = *reinterpret_cast<const float4*>(emb + (size_t)tok * 768 + e4 * 4);
    ushort4 o;
    o.x = f2bf(v.x); o.y = f2bf(v.y); o.z = f2bf(v.z); o.w = f2bf(v.w);
    *reinterpret_cast<ushort4*>(X + (size_t)row * 768 + e4 * 4) = o;
}

// ---------------------------------------------------------------------------
// 2) generic fp32 -> bf16 cast (weights, activations)
// ---------------------------------------------------------------------------
__global__ void __launch_bounds__(256)
cast_f32_bf16(const float* __restrict__ in, unsigned short* __restrict__ out, int n)
{
    int i = blockIdx.x * blockDim.x + threadIdx.x;
    if (i < n) out[i] = f2bf(in[i]);
}

// ---------------------------------------------------------------------------
// 3) Conv-as-implicit-GEMM + bias + relu + max-pool over positions.
//    One block (8 waves) per sentence. The 64x768 bf16 slab (96KB) is staged
//    into LDS once via async DMA; each wave owns 2 N-tiles x 4 M-tiles, so
//    each LDS A-fragment feeds 2 WMMAs and each global B-fragment feeds 4.
//    Valid conv windows end exactly at slab element 64*768-1 for all sz;
//    invalid rows are clamped to Mvalid-1 and masked in the pool epilogue.
//    WMMA operand layouts per CDNA5 ISA 7.12.2:
//      A (16x32 bf16): lane l -> M=l%16; elems e: K = 16*(e/8) + 8*(l/16) + e%8
//      B (32x16 bf16): lane l -> N=l%16; elems e: K = 16*(e/8) + 8*(l/16) + e%8
//      C (16x16 f32) : lane l -> N=l%16; vgpr r -> M = r + 8*(l/16)
// ---------------------------------------------------------------------------
__global__ void __launch_bounds__(256)
conv_wmma_maxpool(const unsigned short* __restrict__ X,   // [512][64*768] bf16
                  const unsigned short* __restrict__ W,   // [256][K] bf16 row-major
                  const float* __restrict__ bias,         // [256]
                  float* __restrict__ para,               // [512][768] fp32
                  int K, int Mvalid, int convOff)
{
    extern __shared__ unsigned short xsh[];               // 64*768 bf16 = 96KB
    const int s    = blockIdx.x;
    const int wave = threadIdx.x >> 5;
    const int lane = threadIdx.x & 31;
    const int hf   = lane >> 4;       // lane half (0/1)
    const int l16  = lane & 15;

    // ---- stage sentence slab to LDS with async global->LDS DMA ----
    {
        const uint4*   gsrc = reinterpret_cast<const uint4*>(X + (size_t)s * (64 * 768));
        const unsigned lbase = __builtin_amdgcn_groupstaticsize();  // dyn LDS base
        for (int i = threadIdx.x; i < 6144; i += 256)               // 6144*16B = 96KB
            async_g2l_b128(lbase + (unsigned)i * 16u, gsrc + i);
        wait_async0();
    }
    __syncthreads();

    const int n0 = wave * 2;                               // this wave's N-tiles
    const unsigned short* w0 = W + (size_t)(n0 * 16 + l16) * K;
    const unsigned short* w1 = w0 + (size_t)16 * K;

    const v8f z8 = {0.f, 0.f, 0.f, 0.f, 0.f, 0.f, 0.f, 0.f};
    v8f acc[4][2];
#pragma unroll
    for (int m = 0; m < 4; ++m) { acc[m][0] = z8; acc[m][1] = z8; }

    for (int kk = 0; kk < K; kk += 32) {
        FragAB fb0, fb1;
        fb0.q[0] = *reinterpret_cast<const uint4*>(w0 + kk + 8 * hf);
        fb0.q[1] = *reinterpret_cast<const uint4*>(w0 + kk + 8 * hf + 16);
        fb1.q[0] = *reinterpret_cast<const uint4*>(w1 + kk + 8 * hf);
        fb1.q[1] = *reinterpret_cast<const uint4*>(w1 + kk + 8 * hf + 16);
        if (kk + 256 < K) {
            __builtin_prefetch(w0 + kk + 256, 0, 0);
            __builtin_prefetch(w1 + kk + 256, 0, 0);
        }
#pragma unroll
        for (int m = 0; m < 4; ++m) {
            int t = m * 16 + l16;                 // conv output position (row)
            if (t > Mvalid - 1) t = Mvalid - 1;   // clamp: stays inside slab
            const unsigned short* xa = xsh + t * 768 + kk + 8 * hf;
            FragAB fa;
            fa.q[0] = *reinterpret_cast<const uint4*>(xa);        // ds_load_b128
            fa.q[1] = *reinterpret_cast<const uint4*>(xa + 16);
            acc[m][0] = __builtin_amdgcn_wmma_f32_16x16x32_bf16(
                false, fa.v, false, fb0.v, (short)0, acc[m][0], false, false);
            acc[m][1] = __builtin_amdgcn_wmma_f32_16x16x32_bf16(
                false, fa.v, false, fb1.v, (short)0, acc[m][1], false, false);
        }
    }

    // epilogue: bias + relu + masked max over positions, all in-wave
#pragma unroll
    for (int j = 0; j < 2; ++j) {
        const float bn = bias[(n0 + j) * 16 + l16];
        float vmax = 0.0f;                        // relu output >= 0
#pragma unroll
        for (int m = 0; m < 4; ++m) {
#pragma unroll
            for (int r = 0; r < 8; ++r) {
                int mm  = m * 16 + hf * 8 + r;
                float v = fmaxf(acc[m][j][r] + bn, 0.0f);
                if (mm < Mvalid) vmax = fmaxf(vmax, v);
            }
        }
        vmax = fmaxf(vmax, __shfl_xor(vmax, 16, 32));   // combine lane halves
        if (hf == 0)
            para[(size_t)s * 768 + convOff + (n0 + j) * 16 + l16] = vmax;
    }
}

// ---------------------------------------------------------------------------
// 4) WMMA GEMM: C[512][256] = A(bf16 rows, optional (t,b)<->(b,t) permute)
//    @ W[256][K]^T. Grid 64, block 256 (8 waves), one 16x16 tile per wave.
// ---------------------------------------------------------------------------
__global__ void __launch_bounds__(256)
wmma_gemm_rowmajor(const unsigned short* __restrict__ A,  // [512][K] bf16
                   const unsigned short* __restrict__ W,  // [256][K] bf16
                   float* __restrict__ C,                 // [512][256]
                   int K, int permute)
{
    const int wave  = threadIdx.x >> 5;
    const int lane  = threadIdx.x & 31;
    const int hf    = lane >> 4;
    const int l16   = lane & 15;
    const int tile  = blockIdx.x * 8 + wave;   // 512 tiles total
    const int ntile = tile & 15;
    const int mtile = tile >> 4;

    const int row = mtile * 16 + l16;                       // GEMM row (t*8+b)
    const int ar  = permute ? ((row & 7) * 64 + (row >> 3)) // para row b*64+t
                            : row;
    const unsigned short* pa = A + (size_t)ar * K;
    const unsigned short* pw = W + (size_t)(ntile * 16 + l16) * K;

    const v8f z8 = {0.f, 0.f, 0.f, 0.f, 0.f, 0.f, 0.f, 0.f};
    v8f acc = z8;
    for (int kk = 0; kk < K; kk += 32) {
        FragAB fa, fb;
        fa.q[0] = *reinterpret_cast<const uint4*>(pa + kk + 8 * hf);
        fa.q[1] = *reinterpret_cast<const uint4*>(pa + kk + 8 * hf + 16);
        fb.q[0] = *reinterpret_cast<const uint4*>(pw + kk + 8 * hf);
        fb.q[1] = *reinterpret_cast<const uint4*>(pw + kk + 8 * hf + 16);
        acc = __builtin_amdgcn_wmma_f32_16x16x32_bf16(
            false, fa.v, false, fb.v, (short)0, acc, false, false);
    }
#pragma unroll
    for (int r = 0; r < 8; ++r) {
        int m = mtile * 16 + hf * 8 + r;
        C[(size_t)m * 256 + ntile * 16 + l16] = acc[r];
    }
}

// ---------------------------------------------------------------------------
// 5) LSTM recurrence, one direction per launch. Block 256; h,c,g state in LDS.
//    gates_in[(t*8+b)*256 + j] = x_t(b) . Wih[j]  (precomputed by GEMM)
//    gate order i,f,g,o (64 each). Output hout[(t*8+b)*128 + outOff + hh].
// ---------------------------------------------------------------------------
__global__ void __launch_bounds__(256)
lstm_rec(const float* __restrict__ gates_in, const float* __restrict__ Whh,
         const float* __restrict__ bih, const float* __restrict__ bhh,
         float* __restrict__ hout, int reverse, int outOff)
{
    __shared__ float hbuf[8][64];
    __shared__ float cbuf[8][64];
    __shared__ float gbuf[8][256];
    const int tid = threadIdx.x;

    for (int p = tid; p < 512; p += 256) { hbuf[p >> 6][p & 63] = 0.f; cbuf[p >> 6][p & 63] = 0.f; }
    __syncthreads();

    const float bsum = bih[tid] + bhh[tid];
    float wrow[64];
#pragma unroll
    for (int k = 0; k < 64; ++k) wrow[k] = Whh[tid * 64 + k];

    for (int s = 0; s < 64; ++s) {
        const int t = reverse ? (63 - s) : s;
        for (int b = 0; b < 8; ++b) {
            float acc = gates_in[(size_t)(t * 8 + b) * 256 + tid] + bsum;
#pragma unroll
            for (int k = 0; k < 64; ++k) acc += hbuf[b][k] * wrow[k];
            gbuf[b][tid] = acc;
        }
        __syncthreads();
        for (int p = tid; p < 512; p += 256) {
            const int b = p >> 6, hh = p & 63;
            const float ig = sigm(gbuf[b][hh]);
            const float fg = sigm(gbuf[b][64 + hh]);
            const float gg = tanhf(gbuf[b][128 + hh]);
            const float og = sigm(gbuf[b][192 + hh]);
            const float c  = fg * cbuf[b][hh] + ig * gg;
            const float h  = og * tanhf(c);
            cbuf[b][hh] = c;
            hbuf[b][hh] = h;
            hout[(size_t)(t * 8 + b) * 128 + outOff + hh] = h;
        }
        __syncthreads();
    }
}

// ---------------------------------------------------------------------------
// 6) Linear layer + CRF forward score + gold score -> scalar loss. One block.
//    N_CLASS=5, SOS=0, EOS=4, NEG=-10000
// ---------------------------------------------------------------------------
__global__ void __launch_bounds__(320)
crf_forward(const float* __restrict__ h2, const float* __restrict__ lin_w,
            const float* __restrict__ lin_b, const float* __restrict__ mask,
            const int* __restrict__ tags, const float* __restrict__ trans_g,
            float* __restrict__ out)
{
    __shared__ float feats[64][8][5];
    __shared__ float tr[25];
    __shared__ float partial[8];
    const int tid = threadIdx.x;
    if (tid < 25) tr[tid] = trans_g[tid];
    __syncthreads();

    // feats = ((h*m) @ lin_w.T + lin_b) * m
    for (int idx = tid; idx < 64 * 8 * 5; idx += 320) {
        int t = idx / 40;
        int b = (idx / 5) & 7;
        int k = idx % 5;
        float m = mask[t * 8 + b];
        const float* hp = h2 + (size_t)(t * 8 + b) * 128;
        float hw = 0.f;
        for (int j = 0; j < 128; ++j) hw += hp[j] * lin_w[k * 128 + j];
        feats[t][b][k] = (hw * m + lin_b[k]) * m;
    }
    __syncthreads();

    if (tid < 8) {
        const int b = tid;
        float alpha[5];
        for (int k = 0; k < 5; ++k) alpha[k] = (k == 0) ? 0.f : -10000.f;
        for (int t = 0; t < 64; ++t) {
            float m = mask[t * 8 + b];
            float na[5];
            for (int k = 0; k < 5; ++k) {
                float mx = -3.4e38f;
                for (int j = 0; j < 5; ++j) mx = fmaxf(mx, alpha[j] + tr[k * 5 + j]);
                float ss = 0.f;
                for (int j = 0; j < 5; ++j) ss += expf(alpha[j] + tr[k * 5 + j] - mx);
                na[k] = mx + logf(ss) + feats[t][b][k];
            }
            for (int k = 0; k < 5; ++k) alpha[k] = na[k] * m + alpha[k] * (1.f - m);
        }
        // forward score: logsumexp(alpha + trans[EOS])
        float mx = -3.4e38f;
        for (int k = 0; k < 5; ++k) mx = fmaxf(mx, alpha[k] + tr[20 + k]);
        float ss = 0.f;
        for (int k = 0; k < 5; ++k) ss += expf(alpha[k] + tr[20 + k] - mx);
        float fwd = mx + logf(ss);
        // gold score
        int   prev = 0;           // SOS
        int   len  = 0;
        float gold = 0.f;
        for (int t = 0; t < 64; ++t) {
            int tg  = tags[t * 8 + b];
            float m = mask[t * 8 + b];
            gold += (feats[t][b][tg] + tr[tg * 5 + prev]) * m;
            prev = tg;
            len += (m != 0.f) ? 1 : 0;
        }
        int last = (len > 0) ? tags[(len - 1) * 8 + b] : 0;
        gold += tr[20 + last];
        partial[b] = fwd - gold;
    }
    __syncthreads();
    if (tid == 0) {
        float s = 0.f;
        for (int b = 0; b < 8; ++b) s += partial[b];
        out[0] = s;
    }
}

// ---------------------------------------------------------------------------
// Orchestration
// ---------------------------------------------------------------------------
extern "C" void kernel_launch(void* const* d_in, const int* in_sizes, int n_in,
                              void* d_out, int out_size, void* d_ws, size_t ws_size,
                              hipStream_t stream)
{
    (void)in_sizes; (void)n_in; (void)out_size; (void)ws_size;

    const int*   input_ids = (const int*)  d_in[0];
    const int*   tags      = (const int*)  d_in[1];
    const float* mask      = (const float*)d_in[2];
    /* d_in[3] = para_len (unused by reference math) */
    const float* emb   = (const float*)d_in[4];
    const float* cw3   = (const float*)d_in[5];
    const float* cb3   = (const float*)d_in[6];
    const float* cw4   = (const float*)d_in[7];
    const float* cb4   = (const float*)d_in[8];
    const float* cw5   = (const float*)d_in[9];
    const float* cb5   = (const float*)d_in[10];
    const float* Wih0f = (const float*)d_in[11];
    const float* Whh0f = (const float*)d_in[12];
    const float* bih0f = (const float*)d_in[13];
    const float* bhh0f = (const float*)d_in[14];
    const float* Wih0b = (const float*)d_in[15];
    const float* Whh0b = (const float*)d_in[16];
    const float* bih0b = (const float*)d_in[17];
    const float* bhh0b = (const float*)d_in[18];
    const float* Wih1f = (const float*)d_in[19];
    const float* Whh1f = (const float*)d_in[20];
    const float* bih1f = (const float*)d_in[21];
    const float* bhh1f = (const float*)d_in[22];
    const float* Wih1b = (const float*)d_in[23];
    const float* Whh1b = (const float*)d_in[24];
    const float* bih1b = (const float*)d_in[25];
    const float* bhh1b = (const float*)d_in[26];
    const float* lin_w = (const float*)d_in[27];
    const float* lin_b = (const float*)d_in[28];
    const float* trans = (const float*)d_in[29];

    char*  ws  = (char*)d_ws;
    size_t off = 0;
    auto alloc = [&](size_t bytes) -> char* {
        char* p = ws + off;
        off = (off + bytes + 255) & ~(size_t)255;
        return p;
    };

    unsigned short* Xbf    = (unsigned short*)alloc((512ull * 64 * 768 + 4096) * 2);
    unsigned short* w3bf   = (unsigned short*)alloc(256ull * 2304 * 2);
    unsigned short* w4bf   = (unsigned short*)alloc(256ull * 3072 * 2);
    unsigned short* w5bf   = (unsigned short*)alloc(256ull * 3840 * 2);
    float*          para   = (float*)         alloc(512ull * 768 * 4);
    unsigned short* parabf = (unsigned short*)alloc(512ull * 768 * 2);
    unsigned short* Wi0fbf = (unsigned short*)alloc(256ull * 768 * 2);
    unsigned short* Wi0bbf = (unsigned short*)alloc(256ull * 768 * 2);
    float*          g0f    = (float*)         alloc(512ull * 256 * 4);
    float*          g0b    = (float*)         alloc(512ull * 256 * 4);
    float*          h1     = (float*)         alloc(512ull * 128 * 4);
    unsigned short* h1bf   = (unsigned short*)alloc(512ull * 128 * 2);
    unsigned short* Wi1fbf = (unsigned short*)alloc(256ull * 128 * 2);
    unsigned short* Wi1bbf = (unsigned short*)alloc(256ull * 128 * 2);
    float*          g1f    = (float*)         alloc(512ull * 256 * 4);
    float*          g1b    = (float*)         alloc(512ull * 256 * 4);
    float*          h2     = (float*)         alloc(512ull * 128 * 4);

    // 1) embedding gather + bf16 cast (100 MB read, bandwidth-bound)
    {
        int total = 32768 * 192;
        gather_embed_bf16<<<(total + 255) / 256, 256, 0, stream>>>(input_ids, emb, Xbf, 32768);
    }
    // 2) weight casts to bf16
    cast_f32_bf16<<<(256 * 2304 + 255) / 256, 256, 0, stream>>>(cw3,   w3bf,   256 * 2304);
    cast_f32_bf16<<<(256 * 3072 + 255) / 256, 256, 0, stream>>>(cw4,   w4bf,   256 * 3072);
    cast_f32_bf16<<<(256 * 3840 + 255) / 256, 256, 0, stream>>>(cw5,   w5bf,   256 * 3840);
    cast_f32_bf16<<<(256 * 768  + 255) / 256, 256, 0, stream>>>(Wih0f, Wi0fbf, 256 * 768);
    cast_f32_bf16<<<(256 * 768  + 255) / 256, 256, 0, stream>>>(Wih0b, Wi0bbf, 256 * 768);
    cast_f32_bf16<<<(256 * 128  + 255) / 256, 256, 0, stream>>>(Wih1f, Wi1fbf, 256 * 128);
    cast_f32_bf16<<<(256 * 128  + 255) / 256, 256, 0, stream>>>(Wih1b, Wi1bbf, 256 * 128);

    // 3) conv banks: async-staged implicit-GEMM WMMA + relu + max-pool
    const size_t slab = 64ull * 768 * 2;   // 96KB dynamic LDS per block
    conv_wmma_maxpool<<<512, 256, slab, stream>>>(Xbf, w3bf, cb3, para, 2304, 62, 0);
    conv_wmma_maxpool<<<512, 256, slab, stream>>>(Xbf, w4bf, cb4, para, 3072, 61, 256);
    conv_wmma_maxpool<<<512, 256, slab, stream>>>(Xbf, w5bf, cb5, para, 3840, 60, 512);

    // 4) layer-0 input projections (rows permuted (b*64+t) -> (t*8+b))
    cast_f32_bf16<<<(512 * 768 + 255) / 256, 256, 0, stream>>>(para, parabf, 512 * 768);
    wmma_gemm_rowmajor<<<64, 256, 0, stream>>>(parabf, Wi0fbf, g0f, 768, 1);
    wmma_gemm_rowmajor<<<64, 256, 0, stream>>>(parabf, Wi0bbf, g0b, 768, 1);

    // 5) layer-0 recurrence (fwd writes cols [0,64), bwd [64,128))
    lstm_rec<<<1, 256, 0, stream>>>(g0f, Whh0f, bih0f, bhh0f, h1, 0, 0);
    lstm_rec<<<1, 256, 0, stream>>>(g0b, Whh0b, bih0b, bhh0b, h1, 1, 64);

    // 6) layer-1 input projections + recurrence
    cast_f32_bf16<<<(512 * 128 + 255) / 256, 256, 0, stream>>>(h1, h1bf, 512 * 128);
    wmma_gemm_rowmajor<<<64, 256, 0, stream>>>(h1bf, Wi1fbf, g1f, 128, 0);
    wmma_gemm_rowmajor<<<64, 256, 0, stream>>>(h1bf, Wi1bbf, g1b, 128, 0);
    lstm_rec<<<1, 256, 0, stream>>>(g1f, Whh1f, bih1f, bhh1f, h2, 0, 0);
    lstm_rec<<<1, 256, 0, stream>>>(g1b, Whh1b, bih1b, bhh1b, h2, 1, 64);

    // 7) linear + CRF forward + gold -> scalar loss
    crf_forward<<<1, 320, 0, stream>>>(h2, lin_w, lin_b, mask, tags, trans, (float*)d_out);
}
